// IPEXGPTJAttention_30279519437194
// MI455X (gfx1250) — matrix operand order repaired and checked by reading, not compiled, not run on hardware
//
#include <hip/hip_runtime.h>

#define D_MODEL 4096
#define SEQ     2048
#define BATCH   2
#define NHEADS  16
#define HDIM    256

typedef __bf16 bf16;
typedef __attribute__((ext_vector_type(16))) __bf16        v16bf;
typedef __attribute__((ext_vector_type(8)))  float         v8f;
typedef __attribute__((ext_vector_type(4)))  unsigned int  u32x4;
typedef __attribute__((ext_vector_type(2)))  unsigned int  u32x2;
typedef __attribute__((ext_vector_type(4)))  float         f32x4;

union Frag { v16bf v; u32x4 q[2]; };

static __device__ __forceinline__ bf16  f2b(float f) { return (bf16)f; }
static __device__ __forceinline__ float b2f(bf16 h)  { return (float)h; }

static __device__ __forceinline__ v8f wmma_bf16(v16bf a, v16bf b, v8f c) {
  return __builtin_amdgcn_wmma_f32_16x16x32_bf16(false, a, false, b,
                                                 (short)0, c, false, false);
}

// 16-element register staging unit: global -> regs -> bf16 LDS
template <typename T> struct Stage;
template <> struct Stage<float> {
  f32x4 r[4];
  __device__ __forceinline__ void load(const float* p) {
#pragma unroll
    for (int i = 0; i < 4; ++i) r[i] = *(const f32x4*)(p + 4 * i);
  }
  __device__ __forceinline__ void store(bf16* d) {
#pragma unroll
    for (int i = 0; i < 4; ++i) {
      union { bf16 h[4]; u32x2 u; } p;
      p.h[0] = f2b(r[i].x); p.h[1] = f2b(r[i].y);
      p.h[2] = f2b(r[i].z); p.h[3] = f2b(r[i].w);
      *(u32x2*)(d + 4 * i) = p.u;
    }
  }
};
template <> struct Stage<bf16> {
  u32x2 r[4];
  __device__ __forceinline__ void load(const bf16* p) {
#pragma unroll
    for (int i = 0; i < 4; ++i) r[i] = *(const u32x2*)(p + 4 * i);
  }
  __device__ __forceinline__ void store(bf16* d) {
#pragma unroll
    for (int i = 0; i < 4; ++i) *(u32x2*)(d + 4 * i) = r[i];
  }
};

// ---------------------------------------------------------------------------
// C[M,N] = A[M,K] * W[N,K]^T   (bf16 WMMA, fp32 accumulate)
// block tile 128x256, K-step 32, double-buffered LDS, 8 waves, wave tile 64x64
// ---------------------------------------------------------------------------
template <typename TA, typename TC>
__global__ __launch_bounds__(256)
void gemm_xwt(const TA* __restrict__ A, const float* __restrict__ W,
              TC* __restrict__ C, int M, int N, int K)
{
  const int LS = 40; // 32 + 8 pad (row stride 80B, 16B-aligned)
  __shared__ __align__(16) bf16 sA[2][128 * 40]; // 20 KB
  __shared__ __align__(16) bf16 sB[2][256 * 40]; // 40 KB

  const int t    = threadIdx.x;
  const int lane = t & 31;
  const int wave = t >> 5;
  const int lr   = lane & 15;
  const int hi   = lane >> 4;
  const int wm   = (wave & 1) * 64;   // 2 waves along M
  const int wn   = (wave >> 1) * 64;  // 4 waves along N
  const long row0 = (long)blockIdx.y * 128;
  const long col0 = (long)blockIdx.x * 256;

  v8f acc[4][4];
#pragma unroll
  for (int i = 0; i < 4; ++i)
#pragma unroll
    for (int j = 0; j < 4; ++j)
#pragma unroll
      for (int e = 0; e < 8; ++e) acc[i][j][e] = 0.0f;

  const int aRow = t >> 1;
  const int aCol = (t & 1) * 16;
  const TA*    aPtr = A + (row0 + aRow) * (long)K + aCol;
  const float* wPtr = W + (col0 + t) * (long)K;

  // prologue: fill buffer 0
  {
    Stage<TA> sa;  sa.load(aPtr);
    Stage<float> sw0, sw1; sw0.load(wPtr); sw1.load(wPtr + 16);
    sa.store(&sA[0][aRow * LS + aCol]);
    sw0.store(&sB[0][t * LS]);
    sw1.store(&sB[0][t * LS + 16]);
  }

  const int nk = K >> 5;
  for (int kk = 0; kk < nk; ++kk) {
    __syncthreads();
    const int cur = kk & 1;
    const bool more = (kk + 1) < nk;

    // 1) issue next tile's global loads early (hidden behind WMMAs)
    Stage<TA> sa;
    Stage<float> sw0, sw1;
    if (more) {
      const int k0 = (kk + 1) << 5;
      sa.load(aPtr + k0);
      sw0.load(wPtr + k0);
      sw1.load(wPtr + k0 + 16);
    }

    // 2) fragment loads from current buffer + 16 WMMAs
    Frag af[4], bfr[4];
#pragma unroll
    for (int mt = 0; mt < 4; ++mt) {
      const bf16* p = &sA[cur][(wm + mt * 16 + lr) * LS + hi * 8];
      af[mt].q[0] = *(const u32x4*)p;
      af[mt].q[1] = *(const u32x4*)(p + 16);
    }
#pragma unroll
    for (int nt = 0; nt < 4; ++nt) {
      const bf16* p = &sB[cur][(wn + nt * 16 + lr) * LS + hi * 8];
      bfr[nt].q[0] = *(const u32x4*)p;
      bfr[nt].q[1] = *(const u32x4*)(p + 16);
    }
#pragma unroll
    for (int mt = 0; mt < 4; ++mt)
#pragma unroll
      for (int nt = 0; nt < 4; ++nt)
        acc[mt][nt] = wmma_bf16(af[mt].v, bfr[nt].v, acc[mt][nt]);

    // 3) convert + store staged tile into the other buffer
    if (more) {
      sa.store(&sA[cur ^ 1][aRow * LS + aCol]);
      sw0.store(&sB[cur ^ 1][t * LS]);
      sw1.store(&sB[cur ^ 1][t * LS + 16]);
    }
  }

#pragma unroll
  for (int mt = 0; mt < 4; ++mt)
#pragma unroll
    for (int nt = 0; nt < 4; ++nt) {
      const long r0 = row0 + wm + mt * 16 + hi * 8;
      const long c  = col0 + wn + nt * 16 + lr;
      TC* op = C + r0 * (long)N + c;
#pragma unroll
      for (int v = 0; v < 8; ++v)
        op[(long)v * N] = (TC)acc[mt][nt][v];
    }
}

// ---------------------------------------------------------------------------
// GPT-J interleaved RoPE on first 64 dims of each head, in place (bf16)
// ---------------------------------------------------------------------------
__global__ __launch_bounds__(256)
void rope_gptj(bf16* __restrict__ Q, bf16* __restrict__ K,
               const int* __restrict__ pos)
{
  const int idx = blockIdx.x * blockDim.x + threadIdx.x;
  const int j = idx & 31;
  const int h = (idx >> 5) & 15;
  const int s = (idx >> 9) & (SEQ - 1);
  const int b = idx >> 20;
  const float p    = (float)pos[s];
  const float invf = __expf(-(float)j * 0.28782313662f); // ln(10000)/32
  const float ang  = p * invf;
  const float sn = __sinf(ang), cs = __cosf(ang);
  const long base = (((long)b * SEQ + s) * D_MODEL) + h * HDIM + 2 * j;
  {
    const float x1 = b2f(Q[base]), x2 = b2f(Q[base + 1]);
    Q[base]     = f2b(x1 * cs - x2 * sn);
    Q[base + 1] = f2b(x2 * cs + x1 * sn);
  }
  {
    const float x1 = b2f(K[base]), x2 = b2f(K[base + 1]);
    K[base]     = f2b(x1 * cs - x2 * sn);
    K[base + 1] = f2b(x2 * cs + x1 * sn);
  }
}

// async global->LDS 16B copy (per lane), offset applies to BOTH addresses
#define ASYNC_CP16(ldsOff, gp, IMM)                                        \
  asm volatile("global_load_async_to_lds_b128 %0, %1, off offset:" #IMM   \
               :: "v"(ldsOff), "v"(gp) : "memory")

// ---------------------------------------------------------------------------
// Causal flash attention, one block = (b, h, 64-query tile), 4 waves
// ---------------------------------------------------------------------------
__global__ __launch_bounds__(128)
void flash_attn(const bf16* __restrict__ Q, const bf16* __restrict__ Kb,
                const bf16* __restrict__ Vb, bf16* __restrict__ Ctx)
{
  __shared__ __align__(16) bf16 sK[64 * 264];    // [key][d], d-padded
  __shared__ __align__(16) bf16 sVt[128 * 72];   // [d(half)][key], key-padded
  __shared__ __align__(16) bf16 sP[4 * 16 * 72]; // per-wave [row][key]

  const int t = threadIdx.x;
  const int lane = t & 31, wave = t >> 5;
  const int lr = lane & 15, hi = lane >> 4;
  const int qt = blockIdx.x;
  const int bh = blockIdx.y;
  const int b = bh >> 4, h = bh & 15;
  const int q0 = qt * 64;
  const long headOff = ((long)b * SEQ) * D_MODEL + (long)h * HDIM;

  // Q fragments (16 rows x 256 d per wave) held in registers
  Frag qf[8];
  {
    const bf16* qp = Q + headOff + (long)(q0 + wave * 16 + lr) * D_MODEL;
#pragma unroll
    for (int dc = 0; dc < 8; ++dc) {
      qf[dc].q[0] = *(const u32x4*)(qp + dc * 32 + hi * 8);
      qf[dc].q[1] = *(const u32x4*)(qp + dc * 32 + hi * 8 + 16);
    }
  }

  v8f ctx[16];
#pragma unroll
  for (int dt = 0; dt < 16; ++dt)
#pragma unroll
    for (int e = 0; e < 8; ++e) ctx[dt][e] = 0.0f;
  float mrow[8], lrow[8];
#pragma unroll
  for (int v = 0; v < 8; ++v) { mrow[v] = -3.0e38f; lrow[v] = 0.0f; }

  const int ldRow = t >> 1;
  const int dbase = (t & 1) * 128;

  for (int kt = 0; kt <= qt; ++kt) {
    const long kvRow = headOff + (long)(kt * 64 + ldRow) * D_MODEL;
    // prefetch next K/V tile into cache (global_prefetch_b8)
    if (kt < qt) {
      const long nRow = headOff + (long)((kt + 1) * 64 + ldRow) * D_MODEL;
      __builtin_prefetch((const void*)(Kb + nRow + dbase), 0, 1);
      __builtin_prefetch((const void*)(Vb + nRow + dbase), 0, 1);
    }
    __syncthreads();
    { // K tile 64x256 bf16: async copy cache->LDS, 256B per thread
      const unsigned ldsK =
          (unsigned)(unsigned long long)(const void*)&sK[ldRow * 264 + dbase];
      const bf16* gk = Kb + kvRow + dbase;
      ASYNC_CP16(ldsK, gk, 0);   ASYNC_CP16(ldsK, gk, 16);
      ASYNC_CP16(ldsK, gk, 32);  ASYNC_CP16(ldsK, gk, 48);
      ASYNC_CP16(ldsK, gk, 64);  ASYNC_CP16(ldsK, gk, 80);
      ASYNC_CP16(ldsK, gk, 96);  ASYNC_CP16(ldsK, gk, 112);
      ASYNC_CP16(ldsK, gk, 128); ASYNC_CP16(ldsK, gk, 144);
      ASYNC_CP16(ldsK, gk, 160); ASYNC_CP16(ldsK, gk, 176);
      ASYNC_CP16(ldsK, gk, 192); ASYNC_CP16(ldsK, gk, 208);
      ASYNC_CP16(ldsK, gk, 224); ASYNC_CP16(ldsK, gk, 240);
    }
    { // V tile half 0 (d 0..127), transposed through VGPRs
      const int dl = (t & 1) * 64;
      const bf16* vp = Vb + kvRow + dl;
#pragma unroll
      for (int i = 0; i < 8; ++i) {
        union { u32x4 u; bf16 hv[8]; } vv;
        vv.u = *(const u32x4*)(vp + 8 * i);
#pragma unroll
        for (int c = 0; c < 8; ++c)
          sVt[(dl + 8 * i + c) * 72 + ldRow] = vv.hv[c];
      }
    }
    asm volatile("s_wait_asynccnt 0x0" ::: "memory");
    __syncthreads();

    // scores: 16 q-rows x 64 keys per wave (batched fragment loads)
    v8f sc[4];
#pragma unroll
    for (int nt = 0; nt < 4; ++nt)
#pragma unroll
      for (int e = 0; e < 8; ++e) sc[nt][e] = 0.0f;
#pragma unroll
    for (int dc = 0; dc < 8; ++dc) {
      Frag kf[4];
#pragma unroll
      for (int nt = 0; nt < 4; ++nt) {
        const bf16* p = &sK[(nt * 16 + lr) * 264 + dc * 32 + hi * 8];
        kf[nt].q[0] = *(const u32x4*)p;
        kf[nt].q[1] = *(const u32x4*)(p + 16);
      }
#pragma unroll
      for (int nt = 0; nt < 4; ++nt)
        sc[nt] = wmma_bf16(qf[dc].v, kf[nt].v, sc[nt]);
    }

    // online softmax (rows split across 16-lane halves)
    float corr[8];
    float sv[4][8];
#pragma unroll
    for (int v = 0; v < 8; ++v) {
      const int qrow = q0 + wave * 16 + v + 8 * hi;
      float tmax = -3.0e38f;
#pragma unroll
      for (int nt = 0; nt < 4; ++nt) {
        const int key = kt * 64 + nt * 16 + lr;
        const float s = (key <= qrow) ? sc[nt][v] * 0.0625f : -4094.0f;
        sv[nt][v] = s;
        tmax = fmaxf(tmax, s);
      }
#pragma unroll
      for (int off = 1; off < 16; off <<= 1)
        tmax = fmaxf(tmax, __shfl_xor(tmax, off, 32));
      const float mnew = fmaxf(mrow[v], tmax);
      corr[v] = __expf(mrow[v] - mnew);
      float rsum = 0.0f;
#pragma unroll
      for (int nt = 0; nt < 4; ++nt) {
        const float pe = __expf(sv[nt][v] - mnew);
        sv[nt][v] = pe;
        rsum += pe;
      }
#pragma unroll
      for (int off = 1; off < 16; off <<= 1)
        rsum += __shfl_xor(rsum, off, 32);
      lrow[v] = lrow[v] * corr[v] + rsum;
      mrow[v] = mnew;
    }
#pragma unroll
    for (int dt = 0; dt < 16; ++dt)
#pragma unroll
      for (int v = 0; v < 8; ++v) ctx[dt][v] *= corr[v];

    // P (C-layout) -> LDS row-major so it can be re-read as A fragments
#pragma unroll
    for (int v = 0; v < 8; ++v)
#pragma unroll
      for (int nt = 0; nt < 4; ++nt)
        sP[(wave * 16 + v + 8 * hi) * 72 + nt * 16 + lr] = f2b(sv[nt][v]);
    __syncthreads();

    Frag pf[2];
#pragma unroll
    for (int kc = 0; kc < 2; ++kc) {
      const bf16* p = &sP[(wave * 16 + lr) * 72 + kc * 32 + hi * 8];
      pf[kc].q[0] = *(const u32x4*)p;
      pf[kc].q[1] = *(const u32x4*)(p + 16);
    }
    // ctx += P * V  (d 0..127), batched V-fragment loads
#pragma unroll
    for (int kc = 0; kc < 2; ++kc)
#pragma unroll
      for (int g = 0; g < 2; ++g) {
        Frag vf[4];
#pragma unroll
        for (int i = 0; i < 4; ++i) {
          const int dt = g * 4 + i;
          const bf16* p = &sVt[(dt * 16 + lr) * 72 + kc * 32 + hi * 8];
          vf[i].q[0] = *(const u32x4*)p;
          vf[i].q[1] = *(const u32x4*)(p + 16);
        }
#pragma unroll
        for (int i = 0; i < 4; ++i)
          ctx[g * 4 + i] = wmma_bf16(pf[kc].v, vf[i].v, ctx[g * 4 + i]);
      }
    __syncthreads();
    { // V tile half 1 (d 128..255), transposed
      const int dl = (t & 1) * 64;
      const bf16* vp = Vb + kvRow + 128 + dl;
#pragma unroll
      for (int i = 0; i < 8; ++i) {
        union { u32x4 u; bf16 hv[8]; } vv;
        vv.u = *(const u32x4*)(vp + 8 * i);
#pragma unroll
        for (int c = 0; c < 8; ++c)
          sVt[(dl + 8 * i + c) * 72 + ldRow] = vv.hv[c];
      }
    }
    __syncthreads();
#pragma unroll
    for (int kc = 0; kc < 2; ++kc)
#pragma unroll
      for (int g = 0; g < 2; ++g) {
        Frag vf[4];
#pragma unroll
        for (int i = 0; i < 4; ++i) {
          const int dt = 8 + g * 4 + i;
          const bf16* p = &sVt[((dt - 8) * 16 + lr) * 72 + kc * 32 + hi * 8];
          vf[i].q[0] = *(const u32x4*)p;
          vf[i].q[1] = *(const u32x4*)(p + 16);
        }
#pragma unroll
        for (int i = 0; i < 4; ++i)
          ctx[8 + g * 4 + i] = wmma_bf16(pf[kc].v, vf[i].v, ctx[8 + g * 4 + i]);
      }
  }

  // normalize and store ctx as [b][s][h*256+d]
#pragma unroll
  for (int v = 0; v < 8; ++v) {
    const float inv = 1.0f / lrow[v];
    bf16* op = Ctx + headOff + (long)(q0 + wave * 16 + v + 8 * hi) * D_MODEL;
#pragma unroll
    for (int dt = 0; dt < 16; ++dt)
      op[dt * 16 + lr] = f2b(ctx[dt][v] * inv);
  }
}

// ---------------------------------------------------------------------------
extern "C" void kernel_launch(void* const* d_in, const int* in_sizes, int n_in,
                              void* d_out, int out_size, void* d_ws, size_t ws_size,
                              hipStream_t stream)
{
  const float* X  = (const float*)d_in[0];
  const float* Wq = (const float*)d_in[1];
  const float* Wk = (const float*)d_in[2];
  const float* Wv = (const float*)d_in[3];
  const float* Wo = (const float*)d_in[4];
  const int*  pos = (const int*)d_in[5];
  float* out = (float*)d_out;

  const size_t planeElems = (size_t)BATCH * SEQ * D_MODEL; // 16M bf16 = 32MB
  bf16* qb = (bf16*)d_ws;
  bf16* kb = qb + planeElems;
  bf16* vb = kb + planeElems;
  bf16* cb = vb + planeElems;

  const int M = BATCH * SEQ;
  dim3 gGrid(D_MODEL / 256, M / 128);

  gemm_xwt<float, bf16><<<gGrid, 256, 0, stream>>>(X, Wq, qb, M, D_MODEL, D_MODEL);
  gemm_xwt<float, bf16><<<gGrid, 256, 0, stream>>>(X, Wk, kb, M, D_MODEL, D_MODEL);
  gemm_xwt<float, bf16><<<gGrid, 256, 0, stream>>>(X, Wv, vb, M, D_MODEL, D_MODEL);

  rope_gptj<<<(BATCH * SEQ * NHEADS * 32) / 256, 256, 0, stream>>>(qb, kb, pos);

  dim3 fGrid(SEQ / 64, BATCH * NHEADS);
  flash_attn<<<fGrid, 128, 0, stream>>>(qb, kb, vb, cb);

  gemm_xwt<bf16, float><<<gGrid, 256, 0, stream>>>(cb, Wo, out, M, D_MODEL, D_MODEL);

  (void)in_sizes; (void)n_in; (void)out_size; (void)ws_size;
}
